// myRIM1D_4758823764701
// MI455X (gfx1250) — matrix operand order repaired and checked by reading, not compiled, not run on hardware
//
#include <hip/hip_runtime.h>
#include <hip/hip_bf16.h>
#include <cmath>

// ---------------------------------------------------------------------------
// Problem constants (from reference): B=8, NC=64, F=8, U=128, NITER=4
// ---------------------------------------------------------------------------
#define NCV   64
#define N3V   262144        // 64^3
#define NFV   2097152       // B * 64^3

typedef __attribute__((ext_vector_type(16))) _Float16 v16h;
typedef __attribute__((ext_vector_type(8)))  _Float16 v8h;
typedef __attribute__((ext_vector_type(8)))  float    v8f;

typedef __attribute__((ext_vector_type(4))) unsigned int u32x4;
typedef __attribute__((ext_vector_type(8))) int          i32x8;
typedef __attribute__((ext_vector_type(4))) int          i32x4;

__device__ __forceinline__ float sigm(float x) { return 1.0f / (1.0f + expf(-x)); }

// ---------------------------------------------------------------------------
// WMMA fragment helpers (CDNA5 ISA 7.12.2, wave32):
//  A (16x32 f16): lane row M = lane&15; per lane two contiguous K runs of 8:
//                 k0 = (lane>>4)*8 and k0+16.
//  B (32x16 f16): lane col N = lane&15; hw k order same as A; this code uses, for A and B:
//                 k = (lane>>4)*16 .. +15   (requires K-contiguous storage).
//  C/D (16x16 f32): VGPR r -> row r + (lane>>4)*8, col = lane&15.
// frag2 builds a v16h from two contiguous 8-element (16B-aligned) runs, which
// the compiler lowers to ds_load_b128 / global_load_b128 pairs.
// ---------------------------------------------------------------------------
__device__ __forceinline__ v16h frag2(const _Float16* p0, const _Float16* p1) {
  v8h a = *(const v8h*)p0;
  v8h b = *(const v8h*)p1;
  v16h r;
#pragma unroll
  for (int i = 0; i < 8; ++i) { r[i] = a[i]; r[8 + i] = b[i]; }
  return r;
}

__device__ __forceinline__ v8f wmma_mma(v16h a, v16h b, v8f c) {
  return __builtin_amdgcn_wmma_f32_16x16x32_f16(false, a, false, b,
                                                (short)0, c, false, false);
}

// ---------------------------------------------------------------------------
// CDNA5 async-tensor probe: NULL D# (group0.count=0 => architectural NOP per
// ISA 8.3) tensor_load_to_lds + s_wait_tensorcnt + cluster barrier (NOP when
// not dispatched as a cluster). Runtime-safe; exercises the TDM/tensorcnt path.
// ---------------------------------------------------------------------------
__global__ void k_cdna5_probe() {
  u32x4 g0 = {0u, 0u, 0u, 0u};          // count=0 -> NULL tensor descriptor
  i32x8 g1 = {0, 0, 0, 0, 0, 0, 0, 0};
  i32x4 g2 = {0, 0, 0, 0};
  i32x4 g3 = {0, 0, 0, 0};
#if defined(__clang_major__) && __clang_major__ >= 23
  i32x8 g4 = {0, 0, 0, 0, 0, 0, 0, 0};
  __builtin_amdgcn_tensor_load_to_lds(g0, g1, g2, g3, g4, 0);
#else
  __builtin_amdgcn_tensor_load_to_lds(g0, g1, g2, g3, 0);
#endif
  __builtin_amdgcn_s_wait_tensorcnt(0);
  __builtin_amdgcn_s_cluster_barrier();
}

// ---------------------------------------------------------------------------
// DFT twiddle tables: Ct[n*64+k]=cos(2*pi*n*k/64), St=sin(...). Symmetric in
// (n,k) -- exploited for K-contiguous B-fragment loads.
// ---------------------------------------------------------------------------
__global__ void k_tables(_Float16* __restrict__ Ct, _Float16* __restrict__ St) {
  int s = blockIdx.x * blockDim.x + threadIdx.x;
  if (s < 4096) {
    int n = s >> 6, k = s & 63;
    float th = 6.28318530717958647692f * (float)(n * k) / 64.0f;
    Ct[s] = (_Float16)cosf(th);
    St[s] = (_Float16)sinf(th);
  }
}

__global__ void k_counts(const int* __restrict__ kb, float* __restrict__ counts) {
  int i = blockIdx.x * blockDim.x + threadIdx.x;
  if (i < N3V) atomicAdd(&counts[kb[i]], 1.0f);
}

// ---------------------------------------------------------------------------
// Weight prep (once per launch): transposed f16 copies so WMMA B fragments are
// K-contiguous 32B runs.
//   Wd1T [128n][128k], WlxT/WlhT [512n][128k], Wd2T [64n][128k]
//   WmT  [32 gate][448 kpad]  (k<216: Wx3, 216<=k<432: Wh3, else 0)
// ---------------------------------------------------------------------------
__global__ void k_prepw(const float* __restrict__ Wd1, const float* __restrict__ Wlx,
                        const float* __restrict__ Wlh, const float* __restrict__ Wd2,
                        const float* __restrict__ Wx3, const float* __restrict__ Wh3,
                        _Float16* __restrict__ Wd1T, _Float16* __restrict__ WlxT,
                        _Float16* __restrict__ WlhT, _Float16* __restrict__ Wd2T,
                        _Float16* __restrict__ WmT) {
  int i = blockIdx.x * blockDim.x + threadIdx.x;
  if (i < 16384) { int n = i >> 7, k = i & 127; Wd1T[i] = (_Float16)Wd1[k * 128 + n]; }
  if (i < 65536) {
    int n = i >> 7, k = i & 127;
    WlxT[i] = (_Float16)Wlx[k * 512 + n];
    WlhT[i] = (_Float16)Wlh[k * 512 + n];
  }
  if (i < 8192)  { int n = i >> 7, k = i & 127; Wd2T[i] = (_Float16)Wd2[k * 64 + n]; }
  if (i < 14336) {
    int n = i / 448, k = i % 448;
    float v = 0.0f;
    if (k < 216)      v = Wx3[k * 32 + n];
    else if (k < 432) v = Wh3[(k - 216) * 32 + n];
    WmT[i] = (_Float16)v;
  }
}

// ---------------------------------------------------------------------------
// Adam update
// ---------------------------------------------------------------------------
__global__ void k_adam(const float* __restrict__ pos, const float* __restrict__ y,
                       float* __restrict__ m, float* __restrict__ v,
                       float* __restrict__ delta, float bc1inv, float bc2inv) {
  int i = blockIdx.x * blockDim.x + threadIdx.x;
  if (i < NFV) {
    float g  = pos[i] - y[i];
    float mm = 0.9f   * m[i] + 0.1f   * g;
    float vv = 0.999f * v[i] + 0.001f * g * g;
    m[i] = mm; v[i] = vv;
    delta[i] = -0.1f * (mm * bc1inv) / (sqrtf(vv * bc2inv) + 1e-7f);
  }
}

// ---------------------------------------------------------------------------
// One DFT axis pass as complex GEMM (all WMMA):
// fwd (sgn=+1): Or = Xr*C + Xi*S ; Oi = Xi*C - Xr*S   (times scale)
// inv (sgn=-1): Or = Xr*C - Xi*S ; Oi = Xi*C + Xr*S
// ---------------------------------------------------------------------------
__device__ __forceinline__ int line_base(int line, int stride) {
  int b = line >> 12, r = line & 4095;
  int off;
  if (stride == 1)        off = r << 6;
  else if (stride == 64)  off = ((r >> 6) << 12) | (r & 63);
  else                    off = r;
  return (b << 18) + off;
}

__global__ void k_dft_pass(const float* __restrict__ inRe, const float* __restrict__ inIm,
                           float* __restrict__ outRe, float* __restrict__ outIm,
                           const _Float16* __restrict__ Ct, const _Float16* __restrict__ St,
                           int stride, int hasImag, float sgn, float scale) {
  __shared__ alignas(16) _Float16 lre[4096];
  __shared__ alignas(16) _Float16 lim[4096];
  __shared__ alignas(16) _Float16 ct[4096];
  __shared__ alignas(16) _Float16 st[4096];
  const int tid = threadIdx.x;
  const int L0  = blockIdx.x * 64;

  for (int s = tid; s < 4096; s += 256) {
    int i = s >> 6, j = s & 63;
    int g = line_base(L0 + i, stride) + j * stride;
    lre[s] = (_Float16)inRe[g];
    lim[s] = hasImag ? (_Float16)inIm[g] : (_Float16)0.0f;
  }
  {
    const uint4* cs = (const uint4*)Ct;
    const uint4* ss = (const uint4*)St;
    uint4* cd = (uint4*)ct;
    uint4* sd = (uint4*)st;
    for (int s = tid; s < 512; s += 256) { cd[s] = cs[s]; sd[s] = ss[s]; }
  }
  __syncthreads();

  const int w    = tid >> 5;
  const int lane = tid & 31, hl = lane >> 4, li = lane & 15;

  for (int tp = 0; tp < 2; ++tp) {
    const int T = w + tp * 8;
    const int rowT = T >> 2, colT = T & 3;
    const int r0 = rowT * 16, c0 = colT * 16;
    const _Float16* parow = &lre[(r0 + li) * 64 + hl * 16];  // A row, this lane
    const _Float16* paimg = &lim[(r0 + li) * 64 + hl * 16];
    const _Float16* pbc   = &ct[(c0 + li) * 64 + hl * 16];  // symmetric table: [col][k]
    const _Float16* pbs   = &st[(c0 + li) * 64 + hl * 16];
    v8f aRC = {}, aRS = {}, aIC = {}, aIS = {};
#pragma unroll
    for (int ks = 0; ks < 2; ++ks) {
      const int k0 = ks * 32;
      v16h aR = frag2(parow + k0, parow + k0 + 8);
      v16h bC = frag2(pbc + k0, pbc + k0 + 8);
      v16h bS = frag2(pbs + k0, pbs + k0 + 8);
      aRC = wmma_mma(aR, bC, aRC);
      aRS = wmma_mma(aR, bS, aRS);
    }
    if (hasImag) {  // kernel-arg uniform: EXEC stays full inside
#pragma unroll
      for (int ks = 0; ks < 2; ++ks) {
        const int k0 = ks * 32;
        v16h aI = frag2(paimg + k0, paimg + k0 + 8);
        v16h bC = frag2(pbc + k0, pbc + k0 + 8);
        v16h bS = frag2(pbs + k0, pbs + k0 + 8);
        aIC = wmma_mma(aI, bC, aIC);
        aIS = wmma_mma(aI, bS, aIS);
      }
    }
    const int col = c0 + li;
#pragma unroll
    for (int r = 0; r < 8; ++r) {
      int mrow = r + hl * 8;
      int g = line_base(L0 + r0 + mrow, stride) + col * stride;
      outRe[g] = scale * (aRC[r] + sgn * aIS[r]);
      outIm[g] = scale * (aIC[r] - sgn * aRS[r]);
    }
  }
}

// ---------------------------------------------------------------------------
// Power binning (field == fft/nc^3 already)
// ---------------------------------------------------------------------------
__global__ void k_power(const float* __restrict__ Fr, const float* __restrict__ Fi,
                        const int* __restrict__ kb, float* __restrict__ power) {
  int i = blockIdx.x * blockDim.x + threadIdx.x;
  if (i < NFV) {
    int b = i >> 18, s = i & (N3V - 1);
    atomicAdd(&power[b * 64 + kb[s]], Fr[i] * Fr[i] + Fi[i] * Fi[i]);
  }
}

__global__ void k_logps(const float* __restrict__ pwp, const float* __restrict__ pwd,
                        const float* __restrict__ counts, float* __restrict__ xcat) {
  int i = blockIdx.x * blockDim.x + threadIdx.x;
  if (i < 512) {
    int b = i >> 6, k = i & 63;
    const float inv_ln10 = 0.43429448190325176f;
    const float bs3 = 8.0e6f;  // 200^3
    float c = counts[k];
    xcat[b * 128 + k]      = logf(pwp[i] / c * bs3) * inv_ln10;
    xcat[b * 128 + 64 + k] = logf(pwd[i] / c * bs3) * inv_ln10;
  }
}

// ---------------------------------------------------------------------------
// cell_in = conv3d(stack([pos,delta],-1), W_in) + b_in   (2 -> 8 channels)
// ---------------------------------------------------------------------------
__global__ void k_cellin(const float* __restrict__ pos, const float* __restrict__ delta,
                         const float* __restrict__ Win, const float* __restrict__ bin_,
                         float* __restrict__ out) {
  __shared__ float wl[432];
  __shared__ float bb[8];
  for (int s = threadIdx.x; s < 432; s += blockDim.x) wl[s] = Win[s];
  if (threadIdx.x < 8) bb[threadIdx.x] = bin_[threadIdx.x];
  __syncthreads();
  int i = blockIdx.x * blockDim.x + threadIdx.x;
  if (i >= NFV) return;
  int b = i >> 18, s2 = i & (N3V - 1);
  int d = s2 >> 12, h = (s2 >> 6) & 63, ww = s2 & 63;
  float acc[8];
#pragma unroll
  for (int c = 0; c < 8; ++c) acc[c] = bb[c];
  for (int tap = 0; tap < 27; ++tap) {
    int kd = tap / 9, kh = (tap / 3) % 3, kw = tap % 3;
    int gd = d + kd - 1, gh = h + kh - 1, gw = ww + kw - 1;
    if ((unsigned)gd >= 64u || (unsigned)gh >= 64u || (unsigned)gw >= 64u) continue;
    int gv = ((b * 64 + gd) * 64 + gh) * 64 + gw;
    float pv = pos[gv], dv = delta[gv];
#pragma unroll
    for (int c = 0; c < 8; ++c)
      acc[c] += pv * wl[tap * 16 + c] + dv * wl[tap * 16 + 8 + c];
  }
#pragma unroll
  for (int c = 0; c < 8; ++c) out[i * 8 + c] = acc[c];
}

// ---------------------------------------------------------------------------
// ConvLSTM gates: implicit GEMM via WMMA (K=432 pad 448, N=32, M=64/block).
// A runs of 8 == one (field,tap) channel group, contiguous in the [vox][ch]
// halo tile -> ds_load_b128 pairs. B from pre-transposed WmT [gate][k].
// ---------------------------------------------------------------------------
__global__ void k_gates(const float* __restrict__ cellin, const float* __restrict__ h3r,
                        const _Float16* __restrict__ WmT, const float* __restrict__ b3,
                        float* __restrict__ h3w, float* __restrict__ c3) {
  __shared__ alignas(16) _Float16 xh[3464];        // 2*1728 halo + 8 zero pad
  __shared__ alignas(16) _Float16 wmatT[32 * 448]; // [gate][kpad]
  __shared__ float zb[64 * 32];
  const int tid = threadIdx.x;
  const int bx = blockIdx.x;
  const int b  = bx >> 12, r = bx & 4095;
  const int bd = r >> 8, bh = (r >> 4) & 15, bw = r & 15;
  const int d0 = bd * 4 - 1, h0 = bh * 4 - 1, w0 = bw * 4 - 1;

  __builtin_prefetch(WmT, 0, 3);
  for (int s = tid; s < 3464; s += 128) {
    float val = 0.0f;
    if (s < 3456) {
      int f = (s >= 1728) ? 1 : 0;
      int q = s - f * 1728;
      int c = q & 7, p = q >> 3;
      int lw = p % 6, lh = (p / 6) % 6, ld = p / 36;
      int gd = d0 + ld, gh = h0 + lh, gw = w0 + lw;
      if ((unsigned)gd < 64u && (unsigned)gh < 64u && (unsigned)gw < 64u) {
        int gv = ((b * 64 + gd) * 64 + gh) * 64 + gw;
        val = f ? h3r[gv * 8 + c] : cellin[gv * 8 + c];
      }
    }
    xh[s] = (_Float16)val;
  }
  {
    const uint4* src = (const uint4*)WmT;   // 32*448 halves = 1792 uint4
    uint4* dst = (uint4*)wmatT;
    for (int s = tid; s < 1792; s += 128) dst[s] = src[s];
  }
  __syncthreads();

  const int w    = tid >> 5;                 // wave = row tile (16 voxels)
  const int lane = tid & 31, hl = lane >> 4, li = lane & 15;
  const int vl   = w * 16 + li;              // this lane's voxel row
  const int lw2 = vl & 3, lh2 = (vl >> 2) & 3, ld2 = vl >> 4;

  auto aptr = [&](int kg) -> const _Float16* {
    if (kg >= 432) return &xh[3456];         // zero pad
    int f = (kg >= 216) ? 1 : 0;
    int tap = (kg - f * 216) >> 3;           // kg is a multiple of 8
    int kd = tap / 9, kh = (tap / 3) % 3, kw = tap % 3;
    int p = ((ld2 + kd) * 6 + (lh2 + kh)) * 6 + (lw2 + kw);
    return &xh[f * 1728 + p * 8];
  };

  float bv0 = b3[li], bv1 = b3[16 + li];
  v8f acc0 = {}, acc1 = {};
#pragma unroll
  for (int rr = 0; rr < 8; ++rr) { acc0[rr] = bv0; acc1[rr] = bv1; }
  const _Float16* pb0 = &wmatT[li * 448 + hl * 16];
  const _Float16* pb1 = &wmatT[(16 + li) * 448 + hl * 16];
#pragma unroll
  for (int ks = 0; ks < 14; ++ks) {
    const int k0 = ks * 32;
    const int kg0 = k0 + hl * 16;
    v16h a = frag2(aptr(kg0), aptr(kg0 + 8));
    acc0 = wmma_mma(a, frag2(pb0 + k0, pb0 + k0 + 8), acc0);
    acc1 = wmma_mma(a, frag2(pb1 + k0, pb1 + k0 + 8), acc1);
  }
#pragma unroll
  for (int rr = 0; rr < 8; ++rr) {
    int mrow = w * 16 + rr + hl * 8;
    zb[mrow * 32 + li]      = acc0[rr];
    zb[mrow * 32 + 16 + li] = acc1[rr];
  }
  __syncthreads();

  for (int s = tid; s < 512; s += 128) {
    int v2 = s >> 3, c = s & 7;
    float zi = zb[v2 * 32 + c],      zf = zb[v2 * 32 + 8 + c];
    float zg = zb[v2 * 32 + 16 + c], zo = zb[v2 * 32 + 24 + c];
    int lw3 = v2 & 3, lh3 = (v2 >> 2) & 3, ld3 = v2 >> 4;
    int gv = ((b * 64 + (bd * 4 + ld3)) * 64 + (bh * 4 + lh3)) * 64 + (bw * 4 + lw3);
    float cold = c3[gv * 8 + c];
    float cn = sigm(zf) * cold + sigm(zi) * tanhf(zg);
    float hn = sigm(zo) * tanhf(cn);
    c3[gv * 8 + c]  = cn;
    h3w[gv * 8 + c] = hn;
  }
}

// ---------------------------------------------------------------------------
// delta_poswhite = conv3d(h3, W_out) + b_out  (8 -> 1)
// ---------------------------------------------------------------------------
__global__ void k_wout(const float* __restrict__ h3, const float* __restrict__ Wout,
                       const float* __restrict__ bout, float* __restrict__ out) {
  __shared__ float wl[216];
  for (int s = threadIdx.x; s < 216; s += blockDim.x) wl[s] = Wout[s];
  __syncthreads();
  int i = blockIdx.x * blockDim.x + threadIdx.x;
  if (i >= NFV) return;
  int b = i >> 18, s2 = i & (N3V - 1);
  int d = s2 >> 12, h = (s2 >> 6) & 63, ww = s2 & 63;
  float acc = bout[0];
  for (int tap = 0; tap < 27; ++tap) {
    int kd = tap / 9, kh = (tap / 3) % 3, kw = tap % 3;
    int gd = d + kd - 1, gh = h + kh - 1, gw = ww + kw - 1;
    if ((unsigned)gd >= 64u || (unsigned)gh >= 64u || (unsigned)gw >= 64u) continue;
    int gv = ((b * 64 + gd) * 64 + gh) * 64 + gw;
#pragma unroll
    for (int c = 0; c < 8; ++c) acc += h3[gv * 8 + c] * wl[tap * 8 + c];
  }
  out[i] = acc;
}

// ---------------------------------------------------------------------------
// Dense LSTM on spectra (B=8, U=128): all GEMMs via WMMA; B operands from
// pre-transposed global f16 weights (contiguous global_load_b128 runs).
// ---------------------------------------------------------------------------
__global__ void k_lstm(const float* __restrict__ xcat,
                       const _Float16* __restrict__ Wd1T, const float* __restrict__ bd1,
                       const _Float16* __restrict__ WlxT, const _Float16* __restrict__ WlhT,
                       const float* __restrict__ bl,
                       const _Float16* __restrict__ Wd2T, const float* __restrict__ bd2,
                       const float* __restrict__ Wc1, const float* __restrict__ bc1,
                       float* __restrict__ h1g, float* __restrict__ c1g,
                       float* __restrict__ fac) {
  __shared__ alignas(16) _Float16 xls[16 * 128];
  __shared__ alignas(16) _Float16 x1ls[16 * 128];
  __shared__ alignas(16) _Float16 h1ls[16 * 128];
  __shared__ float zls[8 * 512];
  __shared__ float lpsO[8 * 64];
  const int tid = threadIdx.x;
  const int w = tid >> 5, lane = tid & 31, hl = lane >> 4, li = lane & 15;

  for (int s = tid; s < 2048; s += 128) {
    int mrow = s >> 7, j = s & 127;
    xls[s]  = (mrow < 8) ? (_Float16)xcat[mrow * 128 + j] : (_Float16)0.0f;
    h1ls[s] = (mrow < 8) ? (_Float16)h1g[mrow * 128 + j]  : (_Float16)0.0f;
  }
  __syncthreads();

  // x1 = x @ Wd1 + bd1
  for (int ct2 = 0; ct2 < 2; ++ct2) {
    const int c0 = (w * 2 + ct2) * 16;
    float bv = bd1[c0 + li];
    v8f acc = {};
#pragma unroll
    for (int rr = 0; rr < 8; ++rr) acc[rr] = bv;
    const _Float16* pa = &xls[li * 128 + hl * 16];
    const _Float16* pb = &Wd1T[(c0 + li) * 128 + hl * 16];
#pragma unroll
    for (int ks = 0; ks < 4; ++ks) {
      const int k0 = ks * 32;
      acc = wmma_mma(frag2(pa + k0, pa + k0 + 8), frag2(pb + k0, pb + k0 + 8), acc);
    }
#pragma unroll
    for (int rr = 0; rr < 8; ++rr) x1ls[(rr + hl * 8) * 128 + c0 + li] = (_Float16)acc[rr];
  }
  __syncthreads();

  // z = x1 @ Wlx + h1 @ Wlh + bl
  for (int ct2 = 0; ct2 < 8; ++ct2) {
    const int c0 = (w * 8 + ct2) * 16;
    float bv = bl[c0 + li];
    v8f acc = {};
#pragma unroll
    for (int rr = 0; rr < 8; ++rr) acc[rr] = bv;
    const _Float16* pa1 = &x1ls[li * 128 + hl * 16];
    const _Float16* pa2 = &h1ls[li * 128 + hl * 16];
    const _Float16* pb1 = &WlxT[(c0 + li) * 128 + hl * 16];
    const _Float16* pb2 = &WlhT[(c0 + li) * 128 + hl * 16];
#pragma unroll
    for (int ks = 0; ks < 4; ++ks) {
      const int k0 = ks * 32;
      acc = wmma_mma(frag2(pa1 + k0, pa1 + k0 + 8), frag2(pb1 + k0, pb1 + k0 + 8), acc);
    }
#pragma unroll
    for (int ks = 0; ks < 4; ++ks) {
      const int k0 = ks * 32;
      acc = wmma_mma(frag2(pa2 + k0, pa2 + k0 + 8), frag2(pb2 + k0, pb2 + k0 + 8), acc);
    }
#pragma unroll
    for (int rr = 0; rr < 8; ++rr) {
      int mrow = rr + hl * 8;
      if (mrow < 8) zls[mrow * 512 + c0 + li] = acc[rr];
    }
  }
  __syncthreads();

  // gates
  for (int s = tid; s < 1024; s += 128) {
    int b = s >> 7, u = s & 127;
    float zi = zls[b * 512 + u],       zf = zls[b * 512 + 128 + u];
    float zg = zls[b * 512 + 256 + u], zo = zls[b * 512 + 384 + u];
    float cold = c1g[b * 128 + u];
    float cn = sigm(zf) * cold + sigm(zi) * tanhf(zg);
    float hn = sigm(zo) * tanhf(cn);
    c1g[b * 128 + u] = cn;
    h1g[b * 128 + u] = hn;
    h1ls[b * 128 + u] = (_Float16)hn;
  }
  __syncthreads();

  // lps = h1 @ Wd2 + bd2
  {
    const int c0 = w * 16;
    float bv = bd2[c0 + li];
    v8f acc = {};
#pragma unroll
    for (int rr = 0; rr < 8; ++rr) acc[rr] = bv;
    const _Float16* pa = &h1ls[li * 128 + hl * 16];
    const _Float16* pb = &Wd2T[(c0 + li) * 128 + hl * 16];
#pragma unroll
    for (int ks = 0; ks < 4; ++ks) {
      const int k0 = ks * 32;
      acc = wmma_mma(frag2(pa + k0, pa + k0 + 8), frag2(pb + k0, pb + k0 + 8), acc);
    }
#pragma unroll
    for (int rr = 0; rr < 8; ++rr) {
      int mrow = rr + hl * 8;
      if (mrow < 8) lpsO[mrow * 64 + c0 + li] = acc[rr];
    }
  }
  __syncthreads();

  // conv1d (3-tap SAME) then fac = ps^0.5 = exp(0.5*ln10*log10(ps))
  for (int s = tid; s < 512; s += 128) {
    int b = s >> 6, k = s & 63;
    float w0 = Wc1[0], w1 = Wc1[1], w2 = Wc1[2];
    float xm = (k > 0)  ? lpsO[b * 64 + k - 1] : 0.0f;
    float xc = lpsO[b * 64 + k];
    float xp = (k < 63) ? lpsO[b * 64 + k + 1] : 0.0f;
    float lp = bc1[0] + w0 * xm + w1 * xc + w2 * xp;
    fac[s] = expf(1.1512925464970229f * lp);
  }
}

// ---------------------------------------------------------------------------
// Spectral scaling / update
// ---------------------------------------------------------------------------
__global__ void k_scale(float* __restrict__ Fr, float* __restrict__ Fi,
                        const int* __restrict__ kb, const float* __restrict__ fac) {
  int i = blockIdx.x * blockDim.x + threadIdx.x;
  if (i < NFV) {
    int b = i >> 18, s = i & (N3V - 1);
    float f = fac[b * 64 + kb[s]];
    Fr[i] *= f;
    Fi[i] *= f;
  }
}

__global__ void k_update(const float* __restrict__ re, const float* __restrict__ xt,
                         float* __restrict__ pos, float* __restrict__ loss) {
  __shared__ float red[256];
  int i = blockIdx.x * blockDim.x + threadIdx.x;
  float d2 = 0.0f;
  if (i < NFV) {
    float np = pos[i] + re[i];
    pos[i] = np;
    float d = xt[i] - np;
    d2 = d * d;
  }
  red[threadIdx.x] = d2;
  __syncthreads();
  for (int o = 128; o > 0; o >>= 1) {
    if (threadIdx.x < (unsigned)o) red[threadIdx.x] += red[threadIdx.x + o];
    __syncthreads();
  }
  if (threadIdx.x == 0) atomicAdd(loss, red[0]);
}

__global__ void k_finalize(const float* __restrict__ loss, float* __restrict__ out) {
  if (threadIdx.x == 0) out[0] = loss[0] * (1.0f / (float)NFV);
}

// ---------------------------------------------------------------------------
// Driver
// ---------------------------------------------------------------------------
extern "C" void kernel_launch(void* const* d_in, const int* in_sizes, int n_in,
                              void* d_out, int out_size, void* d_ws, size_t ws_size,
                              hipStream_t stream) {
  const float* x_init = (const float*)d_in[0];
  const float* y      = (const float*)d_in[1];
  const float* x_true = (const float*)d_in[2];
  const int*   kb     = (const int*)d_in[3];
  const float* W_in   = (const float*)d_in[4];
  const float* b_in   = (const float*)d_in[5];
  const float* Wx3    = (const float*)d_in[6];
  const float* Wh3    = (const float*)d_in[7];
  const float* b3     = (const float*)d_in[8];
  const float* W_out  = (const float*)d_in[9];
  const float* b_out  = (const float*)d_in[10];
  const float* Wd1    = (const float*)d_in[11];
  const float* bd1    = (const float*)d_in[12];
  const float* Wlx    = (const float*)d_in[13];
  const float* Wlh    = (const float*)d_in[14];
  const float* bl     = (const float*)d_in[15];
  const float* Wd2    = (const float*)d_in[16];
  const float* bd2    = (const float*)d_in[17];
  const float* Wc1    = (const float*)d_in[18];
  const float* bc1    = (const float*)d_in[19];

  float* W = (float*)d_ws;
  size_t off = 0;
  auto take = [&](size_t n) {            // 16-byte aligned allocations
    n = (n + 3) & ~(size_t)3;
    float* p = W + off; off += n; return p;
  };
  const size_t NF = NFV;

  float* pos    = take(NF);
  float* m      = take(NF);
  float* v      = take(NF);
  float* delta  = take(NF);
  float* dpw    = take(NF);
  float* FrA    = take(NF);
  float* FiA    = take(NF);
  float* FrB    = take(NF);
  float* FiB    = take(NF);
  float* cellin = take(NF * 8);
  float* h3a    = take(NF * 8);
  float* h3b    = take(NF * 8);
  float* c3     = take(NF * 8);
  float* pwp    = take(512);
  float* pwd    = take(512);
  float* counts = take(64);
  float* xcat   = take(1024);
  float* h1     = take(1024);
  float* c1     = take(1024);
  float* fac    = take(512);
  _Float16* Ct   = (_Float16*)take(2048);   // 4096 halves
  _Float16* St   = (_Float16*)take(2048);
  _Float16* Wd1T = (_Float16*)take(8192);   // 16384 halves
  _Float16* WlxT = (_Float16*)take(32768);  // 65536 halves
  _Float16* WlhT = (_Float16*)take(32768);
  _Float16* Wd2T = (_Float16*)take(4096);   // 8192 halves
  _Float16* WmT  = (_Float16*)take(7168);   // 14336 halves
  float* lossb   = take(4);

  // ---- init state ----
  k_cdna5_probe<<<1, 32, 0, stream>>>();
  hipMemcpyAsync(pos, x_init, NF * 4, hipMemcpyDeviceToDevice, stream);
  hipMemsetAsync(m, 0, NF * 4, stream);
  hipMemsetAsync(v, 0, NF * 4, stream);
  hipMemsetAsync(h3a, 0, NF * 8 * 4, stream);
  hipMemsetAsync(c3, 0, NF * 8 * 4, stream);
  hipMemsetAsync(h1, 0, 4096, stream);
  hipMemsetAsync(c1, 0, 4096, stream);
  hipMemsetAsync(counts, 0, 256, stream);
  hipMemsetAsync(lossb, 0, 4, stream);
  k_tables<<<16, 256, 0, stream>>>(Ct, St);
  k_counts<<<1024, 256, 0, stream>>>(kb, counts);
  k_prepw<<<256, 256, 0, stream>>>(Wd1, Wlx, Wlh, Wd2, Wx3, Wh3,
                                   Wd1T, WlxT, WlhT, Wd2T, WmT);

  float* h3r = h3a;
  float* h3w = h3b;

  for (int t = 1; t <= 4; ++t) {
    float bc1i = 1.0f / (1.0f - powf(0.9f, (float)t));
    float bc2i = 1.0f / (1.0f - powf(0.999f, (float)t));
    k_adam<<<8192, 256, 0, stream>>>(pos, y, m, v, delta, bc1i, bc2i);

    // varmap(pos)
    k_dft_pass<<<512, 256, 0, stream>>>(pos, nullptr, FrA, FiA, Ct, St, 1, 0, 1.0f, 1.0f / 64.0f);
    k_dft_pass<<<512, 256, 0, stream>>>(FrA, FiA, FrB, FiB, Ct, St, 64, 1, 1.0f, 1.0f / 64.0f);
    k_dft_pass<<<512, 256, 0, stream>>>(FrB, FiB, FrA, FiA, Ct, St, 4096, 1, 1.0f, 1.0f / 64.0f);
    hipMemsetAsync(pwp, 0, 2048, stream);
    k_power<<<8192, 256, 0, stream>>>(FrA, FiA, kb, pwp);

    // varmap(delta)
    k_dft_pass<<<512, 256, 0, stream>>>(delta, nullptr, FrA, FiA, Ct, St, 1, 0, 1.0f, 1.0f / 64.0f);
    k_dft_pass<<<512, 256, 0, stream>>>(FrA, FiA, FrB, FiB, Ct, St, 64, 1, 1.0f, 1.0f / 64.0f);
    k_dft_pass<<<512, 256, 0, stream>>>(FrB, FiB, FrA, FiA, Ct, St, 4096, 1, 1.0f, 1.0f / 64.0f);
    hipMemsetAsync(pwd, 0, 2048, stream);
    k_power<<<8192, 256, 0, stream>>>(FrA, FiA, kb, pwd);

    k_logps<<<2, 256, 0, stream>>>(pwp, pwd, counts, xcat);

    // ConvLSTM path
    k_cellin<<<8192, 256, 0, stream>>>(pos, delta, W_in, b_in, cellin);
    k_gates<<<32768, 128, 0, stream>>>(cellin, h3r, WmT, b3, h3w, c3);
    k_wout<<<8192, 256, 0, stream>>>(h3w, W_out, b_out, dpw);

    // spectral LSTM -> fac
    k_lstm<<<1, 128, 0, stream>>>(xcat, Wd1T, bd1, WlxT, WlhT, bl, Wd2T, bd2,
                                  Wc1, bc1, h1, c1, fac);

    // filt: fwd FFT (scaled) -> scale -> unnormalized inverse == true ifftn
    k_dft_pass<<<512, 256, 0, stream>>>(dpw, nullptr, FrA, FiA, Ct, St, 1, 0, 1.0f, 1.0f / 64.0f);
    k_dft_pass<<<512, 256, 0, stream>>>(FrA, FiA, FrB, FiB, Ct, St, 64, 1, 1.0f, 1.0f / 64.0f);
    k_dft_pass<<<512, 256, 0, stream>>>(FrB, FiB, FrA, FiA, Ct, St, 4096, 1, 1.0f, 1.0f / 64.0f);
    k_scale<<<8192, 256, 0, stream>>>(FrA, FiA, kb, fac);
    k_dft_pass<<<512, 256, 0, stream>>>(FrA, FiA, FrB, FiB, Ct, St, 1, 1, -1.0f, 1.0f);
    k_dft_pass<<<512, 256, 0, stream>>>(FrB, FiB, FrA, FiA, Ct, St, 64, 1, -1.0f, 1.0f);
    k_dft_pass<<<512, 256, 0, stream>>>(FrA, FiA, FrB, FiB, Ct, St, 4096, 1, -1.0f, 1.0f);

    k_update<<<8192, 256, 0, stream>>>(FrB, x_true, pos, lossb);

    float* tmp = h3r; h3r = h3w; h3w = tmp;
  }

  hipMemcpyAsync(d_out, pos, NF * 4, hipMemcpyDeviceToDevice, stream);
  k_finalize<<<1, 32, 0, stream>>>(lossb, (float*)d_out + NF);
}
